// FrozenConvMoE_DSFNet_14053132993144
// MI455X (gfx1250) — matrix operand order repaired and checked by reading, not compiled, not run on hardware
//
#include <hip/hip_runtime.h>
#include <hip/hip_bf16.h>

typedef __attribute__((ext_vector_type(16))) _Float16 v16h;
typedef __attribute__((ext_vector_type(8)))  float    v8f;

#define NB      16          // batch
#define CIN     15          // C*T
#define HIN     384
#define KTOT    735         // 15*7*7
#define GOH     192         // gating conv out
#define EOH     96          // expert conv out / head spatial
#define NEXP    3

// ---- WMMA fragment index maps (contiguous k order used here, 16-bit A 16x32 / B 32x16) ----
__device__ __forceinline__ int kmapA(int i, int hi) {
  // A: lane holds row M=lane%16; halves 0..15 -> K 0..15 (+16 if laneHi)
  return i + (hi ? 16 : 0);
}
__device__ __forceinline__ v8f wmma16(v16h a, v16h b, v8f c) {
  return __builtin_amdgcn_wmma_f32_16x16x32_f16(false, a, false, b, (short)0, c,
                                                false, false);
}
// Unpack 16 consecutive u16 LDS-table entries (two b128 loads) into boff[16].
// B half i of lane maps k = kb + i + 16*laneHi -> table entries are contiguous.
__device__ __forceinline__ void load_boff(const unsigned short* tab, int kb,
                                          int hi, int addc, int* boff) {
  const uint4* t4 = (const uint4*)tab;
  const int q = (kb + hi * 16) >> 3;       // uint4 index (16 u16 = 2 x uint4)
  uint4 qa = t4[q];
  uint4 qb = t4[q + 1];
  unsigned int w[8] = {qa.x, qa.y, qa.z, qa.w, qb.x, qb.y, qb.z, qb.w};
#pragma unroll
  for (int j = 0; j < 8; ++j) {
    boff[2 * j]     = (int)(w[j] & 0xffffu) + addc;
    boff[2 * j + 1] = (int)(w[j] >> 16)     + addc;
  }
}

// =====================================================================
// K1: gating conv 7x7 s2 p3 (15->64) + BN + ReLU, implicit GEMM via WMMA
// grid (12,12,16), block 256 (8 waves). Out tile 16x16; patch 37x37.
// =====================================================================
__global__ __launch_bounds__(256)
void gating_conv_kernel(const float* __restrict__ x,
                        const float* __restrict__ W,      // [64][735]
                        const float* __restrict__ gamma,
                        const float* __restrict__ beta,
                        const float* __restrict__ mean,
                        const float* __restrict__ var,
                        _Float16* __restrict__ gout) {    // [16][64][192][192]
  constexpr int STRIDE = 2, TILE = 16, PHW = (TILE - 1) * STRIDE + 7; // 37
  constexpr int PTOT = CIN * PHW * PHW;                   // 20535
  constexpr int ZPAD = PTOT;                              // zero pad start
  constexpr int PADN = (TILE - 1) * STRIDE * PHW + 15 * STRIDE + 1; // row+col base
  __shared__ _Float16 patch[PTOT + PADN];
  __shared__ __align__(16) unsigned short offtab[736];
  const int b   = blockIdx.z;
  const int oy0 = blockIdx.y * TILE, ox0 = blockIdx.x * TILE;
  const int tid = threadIdx.x;

  const float* xb = x + (size_t)b * CIN * HIN * HIN;
  for (int p = tid; p < PTOT + PADN; p += 256) {
    float v = 0.f;
    if (p < PTOT) {
      int c  = p / (PHW * PHW), r = p % (PHW * PHW);
      int py = r / PHW,         px = r % PHW;
      int gy = oy0 * STRIDE - 3 + py;
      int gx = ox0 * STRIDE - 3 + px;
      if (gy >= 0 && gy < HIN && gx >= 0 && gx < HIN)
        v = xb[(size_t)c * HIN * HIN + gy * HIN + gx];
    }
    patch[p] = (_Float16)v;
  }
  // k -> patch-offset table (block-invariant; built once)
  for (int k = tid; k < 736; k += 256) {
    int off = ZPAD;
    if (k < KTOT) {
      int c = k / 49, t = k % 49;
      off = (c * PHW + t / 7) * PHW + (t % 7);
    }
    offtab[k] = (unsigned short)off;
  }
  __syncthreads();

  const int wave = tid >> 5, lane = tid & 31;
  const int lane16 = lane & 15, hi = lane >> 4;
  const int mtile = wave >> 1;       // 0..3 (64 out ch)
  const int nhalf = wave & 1;        // ntiles [nhalf*8, nhalf*8+8)
  const int baseCol = lane16 * STRIDE;

  v8f zero{};
  v8f acc[8];
#pragma unroll
  for (int i = 0; i < 8; ++i) acc[i] = zero;

  const float* wrow = W + (size_t)(mtile * 16 + lane16) * KTOT;
  auto kstep = [&](int kb, bool guardA) {
    v16h afrag;
#pragma unroll
    for (int i = 0; i < 16; ++i) {
      int k = kb + kmapA(i, hi);
      afrag[i] = (!guardA || k < KTOT) ? (_Float16)wrow[k] : (_Float16)0.f;
    }
    int boff[16];
    load_boff(offtab, kb, hi, baseCol, boff);
#pragma unroll
    for (int nt = 0; nt < 8; ++nt) {
      const int rbase = (nhalf * 8 + nt) * STRIDE * PHW;
      v16h bfrag;
#pragma unroll
      for (int i = 0; i < 16; ++i) bfrag[i] = patch[boff[i] + rbase];
      acc[nt] = wmma16(afrag, bfrag, acc[nt]);
    }
  };
  for (int ks = 0; ks < 22; ++ks) {        // 22*32 = 704 <= 735: no A guard
    __builtin_prefetch(wrow + ks * 32 + 32, 0, 3);   // global_prefetch_b8
    kstep(ks * 32, false);
  }
  kstep(704, true);                        // tail: guard A only (B via offtab)

  // epilogue: BN + ReLU -> f16 (per-channel constants hoisted)
#pragma unroll
  for (int v = 0; v < 8; ++v) {
    const int ch = mtile * 16 + v + (hi ? 8 : 0);
    const float sc = gamma[ch] * rsqrtf(var[ch] + 1e-5f);
    const float sh = beta[ch] - mean[ch] * sc;
    _Float16* orow = gout + (((size_t)b * 64 + ch) * GOH + oy0) * GOH + ox0 + lane16;
#pragma unroll
    for (int nt = 0; nt < 8; ++nt) {
      float val = fmaxf(acc[nt][v] * sc + sh, 0.f);
      orow[(size_t)(nhalf * 8 + nt) * GOH] = (_Float16)val;
    }
  }
}

// =====================================================================
// K2: maxpool 3x3 s2 p1 (192->96) + spatial mean. One block per (b,c).
// =====================================================================
__global__ __launch_bounds__(256)
void pool_reduce_kernel(const _Float16* __restrict__ g,
                        float* __restrict__ feat) {       // [16*64]
  const int bc = blockIdx.x;
  const _Float16* gp = g + (size_t)bc * GOH * GOH;
  float sum = 0.f;
  for (int p = threadIdx.x; p < EOH * EOH; p += 256) {
    int i = p / EOH, j = p % EOH;
    float m = -INFINITY;
#pragma unroll
    for (int dy = 0; dy < 3; ++dy)
#pragma unroll
      for (int dx = 0; dx < 3; ++dx) {
        int r = 2 * i - 1 + dy, c = 2 * j - 1 + dx;
        if (r >= 0 && r < GOH && c >= 0 && c < GOH)
          m = fmaxf(m, (float)gp[r * GOH + c]);
      }
    sum += m;
  }
  __shared__ float red[256];
  red[threadIdx.x] = sum;
  __syncthreads();
  for (int s = 128; s > 0; s >>= 1) {
    if (threadIdx.x < s) red[threadIdx.x] += red[threadIdx.x + s];
    __syncthreads();
  }
  if (threadIdx.x == 0) feat[bc] = red[0] * (1.f / (EOH * EOH));
}

// =====================================================================
// K3: logits -> argmax idx[b], aux loss (softmax/density). 1 block.
// =====================================================================
__global__ __launch_bounds__(64)
void gating_head_kernel(const float* __restrict__ feat,
                        const float* __restrict__ fcw,    // [3][64]
                        const float* __restrict__ fcb,    // [3]
                        int* __restrict__ idx,
                        float* __restrict__ aux_out) {
  __shared__ float logits[NB][NEXP];
  __shared__ int   sidx[NB];
  const int t = threadIdx.x;
  if (t < NB * NEXP) {
    int b = t / NEXP, e = t % NEXP;
    float s = fcb[e];
    for (int c = 0; c < 64; ++c) s += feat[b * 64 + c] * fcw[e * 64 + c];
    logits[b][e] = s;
  }
  __syncthreads();
  if (t < NB) {
    int best = 0; float bv = logits[t][0];
    for (int e = 1; e < NEXP; ++e)
      if (logits[t][e] > bv) { bv = logits[t][e]; best = e; }
    sidx[t] = best;
    idx[t]  = best;
  }
  __syncthreads();
  if (t == 0) {
    float density[NEXP] = {0.f, 0.f, 0.f}, proxy[NEXP] = {0.f, 0.f, 0.f};
    for (int b = 0; b < NB; ++b) {
      density[sidx[b]] += 1.f / NB;
      float mx = fmaxf(logits[b][0], fmaxf(logits[b][1], logits[b][2]));
      float e0 = __expf(logits[b][0] - mx);
      float e1 = __expf(logits[b][1] - mx);
      float e2 = __expf(logits[b][2] - mx);
      float inv = 1.f / (e0 + e1 + e2);
      proxy[0] += e0 * inv / NB;
      proxy[1] += e1 * inv / NB;
      proxy[2] += e2 * inv / NB;
    }
    float aux = (density[0] * proxy[0] + density[1] * proxy[1] +
                 density[2] * proxy[2]) * (float)NEXP;
    *aux_out = 0.01f * aux;
  }
}

// =====================================================================
// K4: routed expert conv1 7x7 s4 p3 (15->64) + bias + ReLU -> y1 f16
// grid (12,12,16), block 128 (4 waves). Out tile 8x8; patch 35x35.
// =====================================================================
__global__ __launch_bounds__(128)
void expert_conv1_kernel(const float* __restrict__ x,
                         const float* __restrict__ Wall,  // [E][64][735]
                         const float* __restrict__ Ball,  // [E][64]
                         const int* __restrict__ idx,
                         _Float16* __restrict__ y1) {     // [16][64][96][96]
  constexpr int STRIDE = 4, TILE = 8, PHW = (TILE - 1) * STRIDE + 7; // 35
  constexpr int PTOT = CIN * PHW * PHW;                   // 18375
  constexpr int ZPAD = PTOT;
  constexpr int PADN = (TILE - 1) * STRIDE * PHW + 7 * STRIDE + 1;
  __shared__ _Float16 patch[PTOT + PADN];
  __shared__ __align__(16) unsigned short offtab[736];
  const int b = blockIdx.z;
  const int e = idx[b];
  const float* W    = Wall + (size_t)e * 64 * KTOT;
  const float* bias = Ball + e * 64;
  const int oy0 = blockIdx.y * TILE, ox0 = blockIdx.x * TILE;
  const int tid = threadIdx.x;

  const float* xb = x + (size_t)b * CIN * HIN * HIN;
  for (int p = tid; p < PTOT + PADN; p += 128) {
    float v = 0.f;
    if (p < PTOT) {
      int c  = p / (PHW * PHW), r = p % (PHW * PHW);
      int py = r / PHW,         px = r % PHW;
      int gy = oy0 * STRIDE - 3 + py;
      int gx = ox0 * STRIDE - 3 + px;
      if (gy >= 0 && gy < HIN && gx >= 0 && gx < HIN)
        v = xb[(size_t)c * HIN * HIN + gy * HIN + gx];
    }
    patch[p] = (_Float16)v;
  }
  for (int k = tid; k < 736; k += 128) {
    int off = ZPAD;
    if (k < KTOT) {
      int c = k / 49, t = k % 49;
      off = (c * PHW + t / 7) * PHW + (t % 7);
    }
    offtab[k] = (unsigned short)off;
  }
  __syncthreads();

  const int wave = tid >> 5, lane = tid & 31;
  const int lane16 = lane & 15, hi = lane >> 4;
  const int mtile = wave;            // 0..3
  const int ocol  = lane16 & 7;      // n = nt*16+lane16 -> orow=nt*2+(lane16>>3)
  const int orow0 = lane16 >> 3;
  const int baseCol = ocol * STRIDE;

  v8f zero{};
  v8f acc[4];
#pragma unroll
  for (int i = 0; i < 4; ++i) acc[i] = zero;

  const float* wrow = W + (size_t)(mtile * 16 + lane16) * KTOT;
  auto kstep = [&](int kb, bool guardA) {
    v16h afrag;
#pragma unroll
    for (int i = 0; i < 16; ++i) {
      int k = kb + kmapA(i, hi);
      afrag[i] = (!guardA || k < KTOT) ? (_Float16)wrow[k] : (_Float16)0.f;
    }
    int boff[16];
    load_boff(offtab, kb, hi, baseCol, boff);
#pragma unroll
    for (int nt = 0; nt < 4; ++nt) {
      const int rbase = (nt * 2 + orow0) * STRIDE * PHW;
      v16h bfrag;
#pragma unroll
      for (int i = 0; i < 16; ++i) bfrag[i] = patch[boff[i] + rbase];
      acc[nt] = wmma16(afrag, bfrag, acc[nt]);
    }
  };
  for (int ks = 0; ks < 22; ++ks) {
    __builtin_prefetch(wrow + ks * 32 + 32, 0, 3);
    kstep(ks * 32, false);
  }
  kstep(704, true);

#pragma unroll
  for (int v = 0; v < 8; ++v) {
    const int ch = mtile * 16 + v + (hi ? 8 : 0);
    const float bch = bias[ch];
    _Float16* yrow = y1 + (((size_t)b * 64 + ch) * EOH + oy0) * EOH + ox0 + ocol;
#pragma unroll
    for (int nt = 0; nt < 4; ++nt) {
      float val = fmaxf(acc[nt][v] + bch, 0.f);
      yrow[(size_t)(nt * 2 + orow0) * EOH] = (_Float16)val;
    }
  }
}

// =====================================================================
// K5: fused head: conv3x3 p1 (64->384) + bias + ReLU + grouped 1x1
// (384->6, 3 groups) + bias. y2 tile lives only in WMMA accumulators.
// grid (6,6,16), block 256 (8 waves). Out tile 16x16.
// =====================================================================
__global__ __launch_bounds__(256)
void head_kernel(const _Float16* __restrict__ y1,
                 const float* __restrict__ w1all,  // [E][384][64][3][3]
                 const float* __restrict__ b1all,  // [E][384]
                 const float* __restrict__ w2all,  // [E][6][128]
                 const float* __restrict__ b2all,  // [E][6]
                 const int* __restrict__ idx,
                 float* __restrict__ out) {        // [16][6][96][96]
  constexpr int PH = 18;
  __shared__ _Float16 patch[64 * PH * PH];         // 41472 B
  __shared__ float    part[8][2][256];             // 16384 B (deterministic red.)
  __shared__ __align__(16) unsigned short offtab[576];
  const int b   = blockIdx.z;
  const int e   = idx[b];
  const int ty0 = blockIdx.y * 16, tx0 = blockIdx.x * 16;
  const int tid = threadIdx.x;

  const _Float16* yb = y1 + (size_t)b * 64 * EOH * EOH;
  for (int p = tid; p < 64 * PH * PH; p += 256) {
    int c  = p / (PH * PH), r = p % (PH * PH);
    int py = r / PH,        px = r % PH;
    int gy = ty0 - 1 + py,  gx = tx0 - 1 + px;
    _Float16 v = (_Float16)0.f;
    if (gy >= 0 && gy < EOH && gx >= 0 && gx < EOH)
      v = yb[c * EOH * EOH + gy * EOH + gx];
    patch[p] = v;
  }
  for (int k = tid; k < 576; k += 256) {
    int c = k / 9, t = k % 9;
    offtab[k] = (unsigned short)((c * PH + t / 3) * PH + (t % 3));
  }
  __syncthreads();

  const int wave = tid >> 5, lane = tid & 31;
  const int lane16 = lane & 15, hi = lane >> 4;
  const float* b1 = b1all + e * 384;

  for (int g = 0; g < NEXP; ++g) {
    const int mb = (g * 8 + wave) * 16;            // channel base
    v8f zero{};
    v8f acc[16];
#pragma unroll
    for (int i = 0; i < 16; ++i) acc[i] = zero;

    const float* w1row = w1all + ((size_t)e * 384 + mb + lane16) * 576;
    for (int ks = 0; ks < 18; ++ks) {
      const int kb = ks * 32;
      __builtin_prefetch(w1row + kb + 32, 0, 3);  // global_prefetch_b8
      v16h afrag;
#pragma unroll
      for (int i = 0; i < 16; ++i)
        afrag[i] = (_Float16)w1row[kb + kmapA(i, hi)];
      int boff[16];
      load_boff(offtab, kb, hi, lane16, boff);
#pragma unroll
      for (int nt = 0; nt < 16; ++nt) {           // orow = nt, ocol = lane16
        v16h bfrag;
#pragma unroll
        for (int i = 0; i < 16; ++i) bfrag[i] = patch[boff[i] + nt * PH];
        acc[nt] = wmma16(afrag, bfrag, acc[nt]);
      }
    }

    // bias + ReLU + grouped 1x1 (partial over this wave's 16 channels)
    const float* w2 = w2all + (size_t)(e * 6 + g * 2) * 128;
    float b1c[8], w20[8], w21[8];
#pragma unroll
    for (int v = 0; v < 8; ++v) {
      int ch  = mb + v + (hi ? 8 : 0);
      b1c[v]  = b1[ch];
      int ic  = ch - g * 128;
      w20[v]  = w2[ic];
      w21[v]  = w2[128 + ic];
    }
#pragma unroll
    for (int nt = 0; nt < 16; ++nt) {
      float c0 = 0.f, c1 = 0.f;
#pragma unroll
      for (int v = 0; v < 8; ++v) {
        float y = fmaxf(acc[nt][v] + b1c[v], 0.f);
        c0 += w20[v] * y;
        c1 += w21[v] * y;
      }
      // fold the laneHi half (channels +8) into lanes 0-15 (same column)
      c0 += __shfl_xor(c0, 16, 32);
      c1 += __shfl_xor(c1, 16, 32);
      if (hi == 0) {
        int col = nt * 16 + lane16;
        part[wave][0][col] = c0;
        part[wave][1][col] = c1;
      }
    }
    __syncthreads();
    // deterministic cross-wave sum + write-out for this group's 2 channels
    for (int p = tid; p < 512; p += 256) {
      int o = p >> 8, col = p & 255;
      float s = 0.f;
#pragma unroll
      for (int w = 0; w < 8; ++w) s += part[w][o][col];
      int oc   = g * 2 + o;
      int orow = col >> 4, ocol = col & 15;
      out[(((size_t)b * 6 + oc) * EOH + ty0 + orow) * EOH + tx0 + ocol] =
          s + b2all[e * 6 + oc];
    }
    __syncthreads();   // before part[] is reused by next group
  }
}

// =====================================================================
extern "C" void kernel_launch(void* const* d_in, const int* in_sizes, int n_in,
                              void* d_out, int out_size, void* d_ws, size_t ws_size,
                              hipStream_t stream) {
  const float* x        = (const float*)d_in[0];
  const float* g_conv_w = (const float*)d_in[1];
  const float* g_gamma  = (const float*)d_in[2];
  const float* g_beta   = (const float*)d_in[3];
  const float* g_mean   = (const float*)d_in[4];
  const float* g_var    = (const float*)d_in[5];
  const float* g_fc_w   = (const float*)d_in[6];
  const float* g_fc_b   = (const float*)d_in[7];
  const float* e_conv1_w = (const float*)d_in[8];
  const float* e_conv1_b = (const float*)d_in[9];
  const float* e_head_w1 = (const float*)d_in[10];
  const float* e_head_b1 = (const float*)d_in[11];
  const float* e_head_w2 = (const float*)d_in[12];
  const float* e_head_b2 = (const float*)d_in[13];
  float* outF = (float*)d_out;

  // workspace layout (f16 activations to halve HBM traffic; L2 is 192MB)
  char* ws = (char*)d_ws;
  _Float16* g_act = (_Float16*)ws;                       // 16*64*192*192 f16
  ws += (size_t)NB * 64 * GOH * GOH * sizeof(_Float16);  // 75,497,472 B
  _Float16* y1 = (_Float16*)ws;                          // 16*64*96*96 f16
  ws += (size_t)NB * 64 * EOH * EOH * sizeof(_Float16);  // 18,874,368 B
  float* feat = (float*)ws;  ws += (size_t)NB * 64 * sizeof(float);
  int* idx = (int*)ws;       ws += (size_t)NB * sizeof(int);

  // 1) gating conv + BN + ReLU (WMMA implicit GEMM)
  gating_conv_kernel<<<dim3(12, 12, NB), 256, 0, stream>>>(
      x, g_conv_w, g_gamma, g_beta, g_mean, g_var, g_act);
  // 2) maxpool + spatial mean
  pool_reduce_kernel<<<NB * 64, 256, 0, stream>>>(g_act, feat);
  // 3) logits, top-1 routing, aux loss (last output element)
  gating_head_kernel<<<1, 64, 0, stream>>>(feat, g_fc_w, g_fc_b, idx,
                                           outF + (out_size - 1));
  // 4) routed expert conv1 (WMMA implicit GEMM, per-sample expert weights)
  expert_conv1_kernel<<<dim3(12, 12, NB), 128, 0, stream>>>(
      x, e_conv1_w, e_conv1_b, idx, y1);
  // 5) fused head: conv3x3 + ReLU + grouped 1x1, y2 never hits memory
  head_kernel<<<dim3(6, 6, NB), 256, 0, stream>>>(
      y1, e_head_w1, e_head_b1, e_head_w2, e_head_b2, idx, outF);
}
